// Head_45835890983154
// MI455X (gfx1250) — matrix operand order repaired and checked by reading, not compiled, not run on hardware
//
#include <hip/hip_runtime.h>
#include <hip/hip_bf16.h>

#define B_DIM 16
#define T_DIM 2048
#define HD    64

typedef _Float16 half_t;
typedef __attribute__((ext_vector_type(16))) _Float16 v16h;
typedef __attribute__((ext_vector_type(8)))  _Float16 v8h;
typedef __attribute__((ext_vector_type(8)))  float    v8f;
typedef int v4i_g __attribute__((vector_size(16)));   // matches builtin param type

// Async global->LDS path (gfx1250): use builtins if this toolchain has them.
#if defined(__has_builtin)
#if __has_builtin(__builtin_amdgcn_global_load_async_to_lds_b128)
#define USE_ASYNC_LDS 1
#endif
#endif
#ifndef USE_ASYNC_LDS
#define USE_ASYNC_LDS 0
#endif

// ---------------------------------------------------------------------------
// WMMA helper: D = A(16x32 f16) * B(32x16 f16) + C(16x16 f32)
// ---------------------------------------------------------------------------
__device__ __forceinline__ v8f wmma_f16(v16h a, v16h b, v8f c) {
    return __builtin_amdgcn_wmma_f32_16x16x32_f16(
        /*neg_a=*/false, a, /*neg_b=*/false, b,
        /*c_mod=*/(short)0, c, /*reuse_a=*/false, /*reuse_b=*/false);
}

// ---------------------------------------------------------------------------
// A-matrix fragment (16x32, MxK), ISA 7.12.2 layout:
//   lane L: M = L&15 ; koff = (L>>4)*8
//   elems [0..7]  = K = koff+0..7 ; elems [8..15] = K = 16+koff+0..7
// ---------------------------------------------------------------------------
__device__ __forceinline__ v16h load_frag_a_f16(const half_t* base, int ld, int lane) {
    const int m    = lane & 15;
    const int koff = (lane >> 4) << 3;
    const half_t* p = base + (size_t)m * ld + koff;
    v8h lo = *(const v8h*)(p);
    v8h hi = *(const v8h*)(p + 16);
    return __builtin_shufflevector(lo, hi, 0,1,2,3,4,5,6,7,8,9,10,11,12,13,14,15);
}

__device__ __forceinline__ v16h load_frag_a_f32(const float* base, int ld, int lane) {
    const int m    = lane & 15;
    const int koff = (lane >> 4) << 3;
    const float* p = base + (size_t)m * ld + koff;
    v16h f;
#pragma unroll
    for (int j = 0; j < 8; ++j) f[j] = (_Float16)p[j];
#pragma unroll
    for (int j = 0; j < 8; ++j) f[8 + j] = (_Float16)p[16 + j];
    return f;
}

// ---------------------------------------------------------------------------
// B-matrix fragment (32x16, KxN): lane L: N = L&15 ; koff = (L>>4)*8 ; K = koff+j (j<8), 16+koff+(j-8) (j>=8)
// Source is row-major N-major / K-contiguous (we feed Bᵀ rows).
// ---------------------------------------------------------------------------
__device__ __forceinline__ v16h load_frag_b_f16(const half_t* base, int ld, int lane) {
    const int n  = lane & 15;
    const int kb = (lane >> 4) << 3;
    const half_t* p = base + (size_t)n * ld + kb;
    v8h lo = *(const v8h*)(p);
    v8h hi = *(const v8h*)(p + 16);
    return __builtin_shufflevector(lo, hi, 0,1,2,3,4,5,6,7,8,9,10,11,12,13,14,15);
}

__device__ __forceinline__ v16h load_frag_b_f32(const float* base, int ld, int lane) {
    const int n  = lane & 15;
    const int kb = (lane >> 4) << 3;
    const float* p = base + (size_t)n * ld + kb;
    v16h f;
#pragma unroll
    for (int j = 0; j < 16; ++j) f[j] = (_Float16)p[(j & 7) + ((j >> 3) << 4)];
    return f;
}

// ---------------------------------------------------------------------------
// 16-byte copy: async DMA to LDS when available, else load+ds_store.
// ---------------------------------------------------------------------------
__device__ __forceinline__ void copy16_to_lds(half_t* lds_dst, const half_t* gsrc) {
#if USE_ASYNC_LDS
    __builtin_amdgcn_global_load_async_to_lds_b128(
        (__attribute__((address_space(1))) v4i_g*)gsrc,
        (__attribute__((address_space(3))) v4i_g*)lds_dst,
        0, 0);
#else
    *(v8h*)lds_dst = *(const v8h*)gsrc;
#endif
}

__device__ __forceinline__ void wait_stage() {
#if USE_ASYNC_LDS
#if __has_builtin(__builtin_amdgcn_s_wait_asynccnt)
    __builtin_amdgcn_s_wait_asynccnt(0);
#else
    asm volatile("s_wait_asynccnt 0" ::: "memory");
#endif
#else
    asm volatile("s_wait_dscnt 0" ::: "memory");
#endif
}

// ---------------------------------------------------------------------------
// Stage 1: q/k/v projections.  One wave per 16-row tile (WMMA NT-gemm).
// Qh, Kh row-major [B,T,64] f16 ; Vt transposed [B,64,T] f16.
// ---------------------------------------------------------------------------
__global__ __launch_bounds__(32)
void proj_kernel(const float* __restrict__ x,
                 const float* __restrict__ Wq,
                 const float* __restrict__ Wk,
                 const float* __restrict__ Wv,
                 half_t* __restrict__ Qh,
                 half_t* __restrict__ Kh,
                 half_t* __restrict__ Vt) {
    const int lane = threadIdx.x;
    const int b    = blockIdx.y;
    const int t0   = blockIdx.x << 4;

    const float* xb = x + ((size_t)b * T_DIM + t0) * HD;
    const v16h xa0 = load_frag_a_f32(xb,      HD, lane);   // k = 0..31
    const v16h xa1 = load_frag_a_f32(xb + 32, HD, lane);   // k = 32..63

    const int rbase = (lane >> 4) << 3;
    const int col   = lane & 15;

    const float* Ws[3] = {Wq, Wk, Wv};
#pragma unroll
    for (int wsel = 0; wsel < 3; ++wsel) {
        const float* W = Ws[wsel];
#pragma unroll
        for (int ht = 0; ht < 4; ++ht) {
            const float* wt = W + (size_t)(ht * 16) * HD;
            v16h wf0 = load_frag_b_f32(wt,      HD, lane);
            v16h wf1 = load_frag_b_f32(wt + 32, HD, lane);
            v8f acc = {0.f,0.f,0.f,0.f,0.f,0.f,0.f,0.f};
            acc = wmma_f16(xa0, wf0, acc);
            acc = wmma_f16(xa1, wf1, acc);
            const int h = ht * 16 + col;
#pragma unroll
            for (int r = 0; r < 8; ++r) {
                const int t = t0 + rbase + r;
                const half_t v = (half_t)acc[r];
                if (wsel == 0)      Qh[((size_t)b * T_DIM + t) * HD + h] = v;
                else if (wsel == 1) Kh[((size_t)b * T_DIM + t) * HD + h] = v;
                else                Vt[((size_t)b * HD + h) * T_DIM + t] = v;
            }
        }
    }
}

// ---------------------------------------------------------------------------
// Stage 2: causal flash attention, 4 waves/block (64 query rows).
// K/V tiles staged into LDS once per block, double-buffered via the async
// global->LDS DMA path; WMMA on jb overlaps the copy of jb+1.
// ---------------------------------------------------------------------------
__global__ __launch_bounds__(128)
void attn_kernel(const half_t* __restrict__ Qh,
                 const half_t* __restrict__ Kh,
                 const half_t* __restrict__ Vt,
                 float* __restrict__ out) {
    __shared__ __align__(16) half_t Ktile[2][32 * HD];   // 2 x 4 KB, row stride 64
    __shared__ __align__(16) half_t Vtile[2][HD * 32];   // 2 x 4 KB, row stride 32
    __shared__ float  Sbuf[4][16 * 33];
    __shared__ __align__(16) half_t Pbuf[4][16 * 40];
    __shared__ float  Arow[4][16];
    __shared__ float  Lrow[4][16];

    const int tid  = threadIdx.x;
    const int lane = tid & 31;
    const int w    = tid >> 5;
    const int b    = blockIdx.y;
    const int r0   = blockIdx.x << 6;
    const int q0   = r0 + (w << 4);

    const half_t* Kbase = Kh + (size_t)b * T_DIM * HD;
    const half_t* Vbase = Vt + (size_t)b * HD * T_DIM;

    // cooperative stage of key-block jb into buffer `buf`
    auto stage = [&](int jb, int buf) {
        const int s0 = jb << 5;
        const half_t* ksrc = Kbase + (size_t)s0 * HD;    // 4 KB contiguous
#pragma unroll
        for (int i = 0; i < 2; ++i) {
            const int c = tid + i * 128;                 // chunk = 8 halves
            copy16_to_lds(&Ktile[buf][c * 8], ksrc + c * 8);
        }
#pragma unroll
        for (int i = 0; i < 2; ++i) {
            const int c    = tid + i * 128;
            const int h    = c >> 2;                     // 4 chunks per V row
            const int part = (c & 3) * 8;
            copy16_to_lds(&Vtile[buf][h * 32 + part],
                          Vbase + (size_t)h * T_DIM + s0 + part);
        }
    };

    const half_t* Qb = Qh + ((size_t)b * T_DIM + q0) * HD;
    const v16h qf0 = load_frag_a_f16(Qb,      HD, lane);
    const v16h qf1 = load_frag_a_f16(Qb + 32, HD, lane);

    v8f o[4];
#pragma unroll
    for (int ht = 0; ht < 4; ++ht) o[ht] = (v8f){0.f,0.f,0.f,0.f,0.f,0.f,0.f,0.f};

    float m_i = -1e30f;      // running row max (lanes 0..15, row = lane)
    float l_i = 0.0f;        // running row sum

    const int rbase = (lane >> 4) << 3;
    const int col   = lane & 15;
    const int nkb   = (r0 + 64) >> 5;     // causal key-block count (uniform)

    stage(0, 0);
    wait_stage();
    __syncthreads();

    for (int jb = 0; jb < nkb; ++jb) {
        const int buf = jb & 1;
        const int s0  = jb << 5;

        if (jb + 1 < nkb) stage(jb + 1, buf ^ 1);        // overlap next copy

        // ---- S = (Q Kᵀ) * 0.125, masked, staged to LDS -------------------
#pragma unroll
        for (int nt = 0; nt < 2; ++nt) {
            const half_t* Kt = &Ktile[buf][(nt * 16) * HD];
            v16h kf0 = load_frag_b_f16(Kt,      HD, lane);
            v16h kf1 = load_frag_b_f16(Kt + 32, HD, lane);
            v8f s = {0.f,0.f,0.f,0.f,0.f,0.f,0.f,0.f};
            s = wmma_f16(qf0, kf0, s);
            s = wmma_f16(qf1, kf1, s);
            const int c = s0 + nt * 16 + col;
#pragma unroll
            for (int r = 0; r < 8; ++r) {
                const int row = q0 + rbase + r;
                float val = s[r] * 0.125f;               // scale = 64^-0.5
                if (c > row) val = -1e30f;               // causal mask
                Sbuf[w][(rbase + r) * 33 + nt * 16 + col] = val;
            }
        }
        asm volatile("s_wait_dscnt 0" ::: "memory");

        // ---- online softmax: lanes 0..15 own one row each ----------------
        if (lane < 16) {
            const int rof = lane * 33;
            float mx = -1e30f;
#pragma unroll
            for (int cc = 0; cc < 32; ++cc) mx = fmaxf(mx, Sbuf[w][rof + cc]);
            const float mnew  = fmaxf(m_i, mx);
            const float alpha = __expf(m_i - mnew);
            float sum = 0.f;
#pragma unroll
            for (int cc = 0; cc < 32; ++cc) {
                const float p = __expf(Sbuf[w][rof + cc] - mnew);
                sum += p;
                Pbuf[w][lane * 40 + cc] = (half_t)p;
            }
            l_i = l_i * alpha + sum;
            m_i = mnew;
            Arow[w][lane] = alpha;
        }
        asm volatile("s_wait_dscnt 0" ::: "memory");

        // ---- rescale O by alpha[row] -------------------------------------
        float ar[8];
#pragma unroll
        for (int r = 0; r < 8; ++r) ar[r] = Arow[w][rbase + r];
#pragma unroll
        for (int ht = 0; ht < 4; ++ht)
#pragma unroll
            for (int r = 0; r < 8; ++r) o[ht][r] *= ar[r];

        // ---- O += P · V  (P A-frag from LDS, V columns from LDS tile) ----
        const v16h pf = load_frag_a_f16(&Pbuf[w][0], 40, lane);
#pragma unroll
        for (int ht = 0; ht < 4; ++ht) {
            v16h vf = load_frag_b_f16(&Vtile[buf][(ht * 16) * 32], 32, lane);
            o[ht] = wmma_f16(pf, vf, o[ht]);
        }

        if (jb + 1 < nkb) wait_stage();    // next buffer fully in LDS
        __syncthreads();                   // safe: nkb uniform per block
    }

    // ---- finalize: divide by row sums, store fp32 ------------------------
    if (lane < 16) Lrow[w][lane] = l_i;
    asm volatile("s_wait_dscnt 0" ::: "memory");
    float inv[8];
#pragma unroll
    for (int r = 0; r < 8; ++r) inv[r] = 1.0f / Lrow[w][rbase + r];

#pragma unroll
    for (int ht = 0; ht < 4; ++ht) {
        const int h = ht * 16 + col;
#pragma unroll
        for (int r = 0; r < 8; ++r) {
            const size_t t = (size_t)q0 + rbase + r;
            out[((size_t)b * T_DIM + t) * HD + h] = o[ht][r] * inv[r];
        }
    }
}

// ---------------------------------------------------------------------------
extern "C" void kernel_launch(void* const* d_in, const int* in_sizes, int n_in,
                              void* d_out, int out_size, void* d_ws, size_t ws_size,
                              hipStream_t stream) {
    const float* x  = (const float*)d_in[0];
    const float* Wq = (const float*)d_in[1];
    const float* Wk = (const float*)d_in[2];
    const float* Wv = (const float*)d_in[3];
    float* out = (float*)d_out;

    const size_t elems = (size_t)B_DIM * T_DIM * HD;   // 2M f16 each
    half_t* Qh = (half_t*)d_ws;
    half_t* Kh = Qh + elems;
    half_t* Vt = Kh + elems;

    proj_kernel<<<dim3(T_DIM / 16, B_DIM), 32, 0, stream>>>(x, Wq, Wk, Wv, Qh, Kh, Vt);
    attn_kernel<<<dim3(T_DIM / 64, B_DIM), 128, 0, stream>>>(Qh, Kh, Vt, out);
}
